// SceneGuidedFusion_3118146257680
// MI455X (gfx1250) — matrix operand order repaired and checked by reading, not compiled, not run on hardware
//
#include <hip/hip_runtime.h>
#include <hip/hip_bf16.h>

typedef __attribute__((ext_vector_type(16))) _Float16 v16h;
typedef __attribute__((ext_vector_type(8)))  _Float16 v8h;
typedef __attribute__((ext_vector_type(8)))  float    v8f;
typedef __attribute__((ext_vector_type(4)))  int      v4i;

constexpr int kB  = 1024;
constexpr int kE  = 256;
constexpr int kH  = 4;
constexpr int kDH = 64;
constexpr int kNF = 16384;
constexpr int kNO = 49152;
constexpr int kL  = 192;
constexpr int kN  = kNF + kNO;      // 65536 total nodes
constexpr float kNEG = -1.0e9f;

constexpr int kWPad = 264;          // LDS row stride in halves (528B -> bank-conflict free)

#ifndef __has_builtin
#define __has_builtin(x) 0
#endif

#if __has_builtin(__builtin_amdgcn_global_load_async_to_lds_b128)
#define ASYNC_LDS 1
#else
#define ASYNC_LDS 0
#endif

// ---------------------------------------------------------------------------
// Small utility kernels: counts / positions / scatter map (replaces the dense
// (B, L, E) materialization in the reference — we only ever touch real nodes)
// ---------------------------------------------------------------------------
__global__ void init_kernel(int* counts_f, int* counts_o, int* slot_map, int n_slots) {
  int i = blockIdx.x * blockDim.x + threadIdx.x;
  if (i < n_slots) slot_map[i] = -1;
  if (i < kB) { counts_f[i] = 0; counts_o[i] = 0; }
}

__global__ void hist_kernel(const int* __restrict__ fb, const int* __restrict__ ob,
                            int* counts_f, int* counts_o) {
  int i = blockIdx.x * blockDim.x + threadIdx.x;
  if (i < kNF)      atomicAdd(&counts_f[fb[i]], 1);
  else if (i < kN)  atomicAdd(&counts_o[ob[i - kNF]], 1);
}

__global__ void scan_kernel(const int* __restrict__ counts_f, const int* __restrict__ counts_o,
                            int* start_f, int* start_o) {
  if (blockIdx.x == 0 && threadIdx.x == 0) {
    int sf = 0, so = 0;
    for (int b = 0; b < kB; ++b) {
      start_f[b] = sf; sf += counts_f[b];
      start_o[b] = so; so += counts_o[b];
    }
  }
}

__global__ void scatter_kernel(const int* __restrict__ fb, const int* __restrict__ ob,
                               const int* __restrict__ start_f, const int* __restrict__ start_o,
                               const int* __restrict__ counts_f, int* __restrict__ slot_map) {
  int i = blockIdx.x * blockDim.x + threadIdx.x;
  if (i < kNF) {
    int b = fb[i];
    int pos = i - start_f[b];
    if (pos < kL) slot_map[b * kL + pos] = i;
  } else if (i < kN) {
    int j = i - kNF;
    int b = ob[j];
    int pos = j - start_o[b] + counts_f[b];
    if (pos < kL) slot_map[b * kL + pos] = kNF + j;
  }
}

// ---------------------------------------------------------------------------
// f32 -> f16 conversion
// ---------------------------------------------------------------------------
__global__ void cvt_f32_f16(const float* __restrict__ src, _Float16* __restrict__ dst, int n) {
  for (int i = blockIdx.x * blockDim.x + threadIdx.x; i < n; i += gridDim.x * blockDim.x)
    dst[i] = (_Float16)src[i];
}

__global__ void cvt_nodes(const float* __restrict__ faces, const float* __restrict__ objs,
                          _Float16* __restrict__ dst) {
  const int total = kN * kE;
  for (int i = blockIdx.x * blockDim.x + threadIdx.x; i < total; i += gridDim.x * blockDim.x) {
    float v = (i < kNF * kE) ? faces[i] : objs[i - kNF * kE];
    dst[i] = (_Float16)v;
  }
}

// ---------------------------------------------------------------------------
// WMMA GEMM: C[M x N] = A[M x 256] * W[N x 256]^T  (f16 in, f32 accumulate)
// Wave tile: 16(M) x 64(N), 4 accumulators reusing one A fragment.
// Block = 256 threads = 8 waves stacked in M -> block tile 128 x 64.
// The 64x256 W tile for a block is staged once into LDS (async load-to-LDS
// on CDNA5, ASYNCcnt tracked), then all 8 waves read B fragments via DS ops.
//
// A fragment (16x32 f16, CDNA5 ISA): lane l<16 holds row M=l,
// K = {k0..k0+7, k0+16..k0+23}; lane l>=16 holds K = {k0+8..15, k0+24..31}.
// B fragment (32x16 f16): lane = column; lanes 0-15 hold K={k0..k0+7, k0+16..k0+23},
// lanes 16-31 hold K={k0+8..15, k0+24..31} (as A) (column n of B == row n of W).
// C/D (16x16 f32): VGPR r holds (M = r + 8*(lane>=16), N = lane&15).
// ---------------------------------------------------------------------------
__device__ __forceinline__ v16h load_frag(const _Float16* p0, const _Float16* p1) {
  v8h a0 = *(const v8h*)p0;
  v8h a1 = *(const v8h*)p1;
  return __builtin_shufflevector(a0, a1, 0, 1, 2, 3, 4, 5, 6, 7,
                                         8, 9, 10, 11, 12, 13, 14, 15);
}

#define WMMA_F16(A, Bf, Cacc) \
  __builtin_amdgcn_wmma_f32_16x16x32_f16(false, (A), false, (Bf), (short)0, (Cacc), false, false)

// Stage 64 rows x 256 halves of W (rows n0..n0+63) into LDS with padded rows.
__device__ __forceinline__ void stage_w(const _Float16* __restrict__ W, int n0,
                                        _Float16* sw, int tid) {
#pragma unroll
  for (int i = 0; i < 8; ++i) {
    const int c   = tid + i * 256;      // 2048 16-byte chunks total
    const int row = c >> 5;             // 0..63
    const int kk  = (c & 31) * 8;       // half offset within row
    const _Float16* gp = W + (size_t)(n0 + row) * kE + kk;
    _Float16* lp = sw + row * kWPad + kk;
#if ASYNC_LDS
    __builtin_amdgcn_global_load_async_to_lds_b128(
        (__attribute__((address_space(1))) v4i*)gp,
        (__attribute__((address_space(3))) v4i*)lp, 0, 0);
#else
    *(v8h*)lp = *(const v8h*)gp;
#endif
  }
#if ASYNC_LDS
#if __has_builtin(__builtin_amdgcn_s_wait_asynccnt)
  __builtin_amdgcn_s_wait_asynccnt(0);
#else
  asm volatile("s_wait_asynccnt 0x0" ::: "memory");
#endif
#endif
  __syncthreads();
}

__global__ __launch_bounds__(256)
void gemm_f16out(const _Float16* __restrict__ A, const _Float16* __restrict__ W,
                 _Float16* __restrict__ C, int ldc) {
  __shared__ alignas(16) _Float16 sw[64 * kWPad];
  const int tid  = threadIdx.x;
  const int lane = tid & 31;
  const int wave = tid >> 5;
  const int l15  = lane & 15;
  const int hi   = lane >> 4;
  const int m0 = blockIdx.x * 128 + wave * 16;
  const int n0 = blockIdx.y * 64;

  stage_w(W, n0, sw, tid);

  const int ka = hi * 8;    // A interleave offset
  const int kb = hi * 8;    // B half-wave K offset
  const _Float16* arow = A + (size_t)(m0 + l15) * kE;

  v8f acc0 = {}, acc1 = {}, acc2 = {}, acc3 = {};
#pragma unroll
  for (int k0 = 0; k0 < kE; k0 += 32) {
    v16h a = load_frag(arow + k0 + ka, arow + k0 + 16 + ka);
    const _Float16* wl = sw + (size_t)l15 * kWPad + k0 + kb;
    v16h b0 = load_frag(wl,              wl + 16);
    v16h b1 = load_frag(wl + 16 * kWPad, wl + 16 * kWPad + 16);
    v16h b2 = load_frag(wl + 32 * kWPad, wl + 32 * kWPad + 16);
    v16h b3 = load_frag(wl + 48 * kWPad, wl + 48 * kWPad + 16);
    acc0 = WMMA_F16(a, b0, acc0);
    acc1 = WMMA_F16(a, b1, acc1);
    acc2 = WMMA_F16(a, b2, acc2);
    acc3 = WMMA_F16(a, b3, acc3);
  }
  const int mr = m0 + hi * 8;
#define STORE16(ACC, T)                                                   \
  { _Float16* cp = C + (size_t)mr * ldc + n0 + (T) * 16 + l15;            \
    _Pragma("unroll") for (int r = 0; r < 8; ++r)                         \
      cp[(size_t)r * ldc] = (_Float16)(ACC)[r]; }
  STORE16(acc0, 0) STORE16(acc1, 1) STORE16(acc2, 2) STORE16(acc3, 3)
#undef STORE16
}

__global__ __launch_bounds__(256)
void gemm_f32out(const _Float16* __restrict__ A, const _Float16* __restrict__ W,
                 float* __restrict__ C, const float* __restrict__ bias,
                 const float* __restrict__ residual) {
  __shared__ alignas(16) _Float16 sw[64 * kWPad];
  const int tid  = threadIdx.x;
  const int lane = tid & 31;
  const int wave = tid >> 5;
  const int l15  = lane & 15;
  const int hi   = lane >> 4;
  const int m0 = blockIdx.x * 128 + wave * 16;
  const int n0 = blockIdx.y * 64;

  stage_w(W, n0, sw, tid);

  const int ka = hi * 8;
  const int kb = hi * 8;
  const _Float16* arow = A + (size_t)(m0 + l15) * kE;

  v8f acc0 = {}, acc1 = {}, acc2 = {}, acc3 = {};
#pragma unroll
  for (int k0 = 0; k0 < kE; k0 += 32) {
    v16h a = load_frag(arow + k0 + ka, arow + k0 + 16 + ka);
    const _Float16* wl = sw + (size_t)l15 * kWPad + k0 + kb;
    v16h b0 = load_frag(wl,              wl + 16);
    v16h b1 = load_frag(wl + 16 * kWPad, wl + 16 * kWPad + 16);
    v16h b2 = load_frag(wl + 32 * kWPad, wl + 32 * kWPad + 16);
    v16h b3 = load_frag(wl + 48 * kWPad, wl + 48 * kWPad + 16);
    acc0 = WMMA_F16(a, b0, acc0);
    acc1 = WMMA_F16(a, b1, acc1);
    acc2 = WMMA_F16(a, b2, acc2);
    acc3 = WMMA_F16(a, b3, acc3);
  }
  const int mr = m0 + hi * 8;
#define STORE32(ACC, T)                                                   \
  { const int col = n0 + (T) * 16 + l15;                                  \
    _Pragma("unroll") for (int r = 0; r < 8; ++r) {                       \
      const int rowi = mr + r;                                            \
      float v = (ACC)[r] + bias[col];                                     \
      if (residual) v += residual[(size_t)rowi * kE + col];               \
      C[(size_t)rowi * kE + col] = v; } }
  STORE32(acc0, 0) STORE32(acc1, 1) STORE32(acc2, 2) STORE32(acc3, 3)
#undef STORE32
}

// ---------------------------------------------------------------------------
// Attention: one block per batch element. fp32 scores/softmax, gathered K/V
// rows (f16) through slot_map. Also emits head-averaged attn_weights.
// ---------------------------------------------------------------------------
__global__ __launch_bounds__(256)
void attn_kernel(const float* __restrict__ qbuf, const _Float16* __restrict__ KV,
                 const int* __restrict__ slot_map, float* __restrict__ aw_out,
                 _Float16* __restrict__ ctx16) {
  const int b = blockIdx.x;
  const int tid = threadIdx.x;
  __shared__ float sq[kE];
  __shared__ float sp[kH * kL];
  __shared__ int   sslot[kL];
  __shared__ float red[256];

  sq[tid] = qbuf[(size_t)b * kE + tid];
  if (tid < kL) sslot[tid] = slot_map[b * kL + tid];
  __syncthreads();

  // scores = q . k / sqrt(DH), masked
  for (int pair = tid; pair < kH * kL; pair += 256) {
    const int h = pair / kL;
    const int l = pair - h * kL;
    const int node = sslot[l];
    float s = kNEG;
    if (node >= 0) {
      const v8h* kr = (const v8h*)(KV + (size_t)node * (2 * kE) + h * kDH);
      float acc = 0.f;
#pragma unroll
      for (int c = 0; c < 8; ++c) {
        v8h kk = kr[c];
#pragma unroll
        for (int j = 0; j < 8; ++j) acc += (float)kk[j] * sq[h * kDH + c * 8 + j];
      }
      s = acc * 0.125f;   // 1/sqrt(64)
    }
    sp[pair] = s;
  }
  __syncthreads();

  // softmax per head over L
  for (int h = 0; h < kH; ++h) {
    float v = (tid < kL) ? sp[h * kL + tid] : kNEG;
    red[tid] = v; __syncthreads();
    for (int off = 128; off > 0; off >>= 1) {
      if (tid < off) red[tid] = fmaxf(red[tid], red[tid + off]);
      __syncthreads();
    }
    const float m = red[0]; __syncthreads();
    const float e = (tid < kL) ? __expf(v - m) : 0.f;
    red[tid] = e; __syncthreads();
    for (int off = 128; off > 0; off >>= 1) {
      if (tid < off) red[tid] += red[tid + off];
      __syncthreads();
    }
    const float inv = 1.f / red[0]; __syncthreads();
    if (tid < kL) sp[h * kL + tid] = e * inv;
    __syncthreads();
  }

  if (tid < kL) {
    float aw = 0.25f * (sp[tid] + sp[kL + tid] + sp[2 * kL + tid] + sp[3 * kL + tid]);
    aw_out[(size_t)b * kL + tid] = aw;
  }

  // ctx[h, d] = sum_l p[h,l] * V[l, h, d]
  const int h = tid >> 6;
  const int d = tid & 63;
  float acc = 0.f;
  for (int l = 0; l < kL; ++l) {
    const int node = sslot[l];
    if (node >= 0)
      acc += sp[h * kL + l] * (float)KV[(size_t)node * (2 * kE) + kE + h * kDH + d];
  }
  ctx16[(size_t)b * kE + tid] = (_Float16)acc;
}

// ---------------------------------------------------------------------------
// LayerNorm over E=256 (one block per row)
// ---------------------------------------------------------------------------
__global__ __launch_bounds__(256)
void ln_kernel(const float* __restrict__ x, const float* __restrict__ gamma,
               const float* __restrict__ beta, float* __restrict__ out) {
  const int b = blockIdx.x, tid = threadIdx.x;
  __shared__ float red[256];
  const float v = x[(size_t)b * kE + tid];
  red[tid] = v; __syncthreads();
  for (int off = 128; off > 0; off >>= 1) { if (tid < off) red[tid] += red[tid + off]; __syncthreads(); }
  const float mu = red[0] * (1.f / kE); __syncthreads();
  const float c = v - mu;
  red[tid] = c * c; __syncthreads();
  for (int off = 128; off > 0; off >>= 1) { if (tid < off) red[tid] += red[tid + off]; __syncthreads(); }
  const float inv = rsqrtf(red[0] * (1.f / kE) + 1e-5f); __syncthreads();
  out[(size_t)b * kE + tid] = c * inv * gamma[tid] + beta[tid];
}

// ---------------------------------------------------------------------------
extern "C" void kernel_launch(void* const* d_in, const int* in_sizes, int n_in,
                              void* d_out, int out_size, void* d_ws, size_t ws_size,
                              hipStream_t stream) {
  (void)in_sizes; (void)n_in; (void)out_size; (void)ws_size;
  const float* scene = (const float*)d_in[0];
  const float* faces = (const float*)d_in[1];
  const float* objs  = (const float*)d_in[2];
  const int*   fb    = (const int*)d_in[3];
  const int*   ob    = (const int*)d_in[4];
  const float* in_w  = (const float*)d_in[5];
  const float* in_b  = (const float*)d_in[6];
  const float* out_w = (const float*)d_in[7];
  const float* out_b = (const float*)d_in[8];
  const float* gamma = (const float*)d_in[9];
  const float* beta  = (const float*)d_in[10];

  char* ws = (char*)d_ws;
  size_t off = 0;
  auto alloc = [&](size_t bytes) -> char* {
    char* p = ws + off;
    off += (bytes + 255) & ~(size_t)255;
    return p;
  };
  _Float16* nodes16 = (_Float16*)alloc((size_t)kN * kE * 2);        // 32 MB
  _Float16* kv16    = (_Float16*)alloc((size_t)kN * 2 * kE * 2);    // 64 MB: [K|V] per node
  _Float16* w16in   = (_Float16*)alloc((size_t)3 * kE * kE * 2);    // wq|wk|wv f16
  _Float16* w16out  = (_Float16*)alloc((size_t)kE * kE * 2);
  _Float16* scene16 = (_Float16*)alloc((size_t)kB * kE * 2);
  float*    qbuf    = (float*)alloc((size_t)kB * kE * 4);
  _Float16* ctx16   = (_Float16*)alloc((size_t)kB * kE * 2);
  float*    xbuf    = (float*)alloc((size_t)kB * kE * 4);
  int* counts_f = (int*)alloc((size_t)kB * 4);
  int* counts_o = (int*)alloc((size_t)kB * 4);
  int* start_f  = (int*)alloc((size_t)kB * 4);
  int* start_o  = (int*)alloc((size_t)kB * 4);
  int* slot_map = (int*)alloc((size_t)kB * kL * 4);

  float* out_fused = (float*)d_out;                     // (B, E)
  float* out_aw    = out_fused + (size_t)kB * kE;       // (B, 1, L)

  // 1) slot assignment
  init_kernel<<<(kB * kL + 255) / 256, 256, 0, stream>>>(counts_f, counts_o, slot_map, kB * kL);
  hist_kernel<<<(kN + 255) / 256, 256, 0, stream>>>(fb, ob, counts_f, counts_o);
  scan_kernel<<<1, 32, 0, stream>>>(counts_f, counts_o, start_f, start_o);
  scatter_kernel<<<(kN + 255) / 256, 256, 0, stream>>>(fb, ob, start_f, start_o, counts_f, slot_map);

  // 2) f16 conversions
  cvt_nodes<<<8192, 256, 0, stream>>>(faces, objs, nodes16);
  cvt_f32_f16<<<768, 256, 0, stream>>>(in_w, w16in, 3 * kE * kE);
  cvt_f32_f16<<<256, 256, 0, stream>>>(out_w, w16out, kE * kE);
  cvt_f32_f16<<<1024, 256, 0, stream>>>(scene, scene16, kB * kE);

  // 3) fused K|V projection over all real nodes: 65536 x 512 (wk,wv are rows
  //    256..767 of in_proj_w, so output col n maps to weight row 256+n).
  gemm_f16out<<<dim3(kN / 128, 512 / 64), 256, 0, stream>>>(
      nodes16, w16in + (size_t)kE * kE, kv16, 2 * kE);

  // 4) Q projection (+bias)
  gemm_f32out<<<dim3(kB / 128, kE / 64), 256, 0, stream>>>(
      scene16, w16in, qbuf, in_b, nullptr);

  // 5) masked attention + attn_weights output
  attn_kernel<<<kB, 256, 0, stream>>>(qbuf, kv16, slot_map, out_aw, ctx16);

  // 6) out projection with bias + residual fused in epilogue
  gemm_f32out<<<dim3(kB / 128, kE / 64), 256, 0, stream>>>(
      ctx16, w16out, xbuf, out_b, scene);

  // 7) LayerNorm -> fused output
  ln_kernel<<<kB, 256, 0, stream>>>(xbuf, gamma, beta, out_fused);
}
